// AGNO_91250875171368
// MI455X (gfx1250) — matrix operand order repaired and checked by reading, not compiled, not run on hardware
//
#include <hip/hip_runtime.h>

typedef __attribute__((ext_vector_type(16))) _Float16     v16h;
typedef __attribute__((ext_vector_type(8)))  float        v8f;
typedef __attribute__((ext_vector_type(4)))  unsigned int v4u;
typedef __attribute__((ext_vector_type(2)))  __fp16       v2h;

#define DEG   16
#define DFEAT 64
#define HID   128

union FragU {
    v4u          u4[2];
    unsigned int u[8];
    v16h         h;
    _Float16     f[16];
};

static __device__ __forceinline__ v16h frag_from(v4u lo, v4u hi4) {
    FragU f; f.u4[0] = lo; f.u4[1] = hi4; return f.h;
}
static __device__ __forceinline__ v8f accf_from(v4u lo, v4u hi4) {
    union { v4u u[2]; v8f f; } c; c.u[0] = lo; c.u[1] = hi4; return c.f;
}
static __device__ __forceinline__ v16h load_frag_u32(const unsigned int* p) {
    FragU f;
    f.u4[0] = *(const v4u*)(p);
    f.u4[1] = *(const v4u*)(p + 4);
    return f.h;
}
static __device__ __forceinline__ v16h load_frag_gl(const _Float16* p) {
    FragU f;
    f.u4[0] = *(const v4u*)(p);
    f.u4[1] = *(const v4u*)(p + 16);
    return f.h;
}
static __device__ __forceinline__ unsigned int pack_rtz(float x, float y) {
    union { v2h h; unsigned int u; } c;
    c.h = __builtin_amdgcn_cvt_pkrtz(x, y);   // v_cvt_pk_rtz_f16_f32
    return c.u;
}
static __device__ __forceinline__ unsigned int lds_off32(const void* p) {
    // flat LDS-aperture address keeps the LDS byte offset in addr[31:0]
    return (unsigned int)(unsigned long long)p;
}

// ---------------------------------------------------------------------------
// Prep 1: y (f32) -> y_h (f16), packed pairs
// ---------------------------------------------------------------------------
__global__ void k_cvt_y(const float* __restrict__ y, unsigned int* __restrict__ y_h,
                        int npairs) {
    int t = blockIdx.x * blockDim.x + threadIdx.x;
    if (t >= npairs) return;
    float2 f = *(const float2*)(y + 2 * t);
    y_h[t] = pack_rtz(f.x, f.y);
}

// ---------------------------------------------------------------------------
// Prep 2: pack W1^T / W2^T into per-lane WMMA A-fragment order (f16 pairs).
// A-fragment (16x32 f16, wave32): lane holds row i = lane&15; VGPR v holds
// K = ((v&4)?16:0) + 8*(lane>>4) + 2*(v&3) + {0,1}.
// w1p[((mt*4+kt)*32 + lane)*8 + v], A = W1^T so element = W1[32kt+k2][16mt+i].
// ---------------------------------------------------------------------------
__global__ void k_pack_w(const float* __restrict__ W1, const float* __restrict__ W2,
                         unsigned int* __restrict__ w1p, unsigned int* __restrict__ w2p) {
    int t = blockIdx.x * blockDim.x + threadIdx.x;
    int v = t & 7, lane = (t >> 3) & 31;
    int i = lane & 15, hi = lane >> 4;
    int k2 = ((v & 4) ? 16 : 0) + 8 * hi + 2 * (v & 3);
    if (t < 8192) {                       // W1^T fragments: 8 mt x 4 kt
        int fk = t >> 8, mt = fk >> 2, kt = fk & 3;
        int r = 32 * kt + k2, c = 16 * mt + i;
        w1p[t] = pack_rtz(W1[r * HID + c], W1[(r + 1) * HID + c]);
    } else if (t < 12288) {               // W2^T fragments: 4 mt x 4 kt
        int u2 = t - 8192;
        int fk = u2 >> 8, mt = fk >> 2, kt = fk & 3;
        int r = 32 * kt + k2, c = 16 * mt + i;
        w2p[u2] = pack_rtz(W2[r * DFEAT + c], W2[(r + 1) * DFEAT + c]);
    }
}

// ---------------------------------------------------------------------------
// Main fused kernel: one wave32 per node (16 edges = one WMMA N-tile).
//   hT(128x16) = W1^T @ aggT   (32 wmma),  relu + softmax column scale,
//   mT(64x16)  = W2^T @ hT     (16 wmma),  butterfly-reduce over edges.
// W1 fragments are explicitly register-resident (256 VGPRs, preloaded once
// per wave). W2 fragments + b1 stay in LDS and are fetched per node through
// asm-volatile ds_load_b128 batches so the compiler cannot promote them
// (which previously overflowed the VGPR budget and spilled to scratch).
// ---------------------------------------------------------------------------
__global__ __launch_bounds__(256, 1) void k_agno(
    const float* __restrict__ y, const long long* __restrict__ indices,
    const float* __restrict__ b1, const float* __restrict__ b2,
    const _Float16* __restrict__ y_h,
    const unsigned int* __restrict__ w1p, const unsigned int* __restrict__ w2p,
    float* __restrict__ out, int n_nodes)
{
    __shared__ unsigned int sW2[4096];   // W2^T A-fragments, 16 KB
    __shared__ float        sB1[HID];    // 512 B

    const int tid = threadIdx.x;
    for (int i = tid; i < 4096; i += 256) sW2[i] = w2p[i];
    if (tid < HID) sB1[tid] = b1[tid];
    __syncthreads();

    const int lane = tid & 31;
    const int col  = lane & 15;          // edge index within node (both halves dup)
    const int hi   = lane >> 4;
    const int nWaves = (int)(gridDim.x * 8);
    int node = blockIdx.x * 8 + (tid >> 5);

    // ---- W1^T fragments -> registers (64 x global_load_b128, L2-resident)
    v16h A1[32];
    #pragma unroll
    for (int fk = 0; fk < 32; ++fk)
        A1[fk] = load_frag_u32(w1p + (fk * 32 + lane) * 8);

    const unsigned w2lds = lds_off32(&sW2[0]);
    const unsigned b1lds = lds_off32(&sB1[0]) + (unsigned)(hi * 32);

    for (; node < n_nodes; node += nWaves) {
        // prefetch next iteration's index row + self feature row
        __builtin_prefetch(indices + (long long)(node + nWaves) * DEG, 0, 0);
        __builtin_prefetch(y_h + (long long)(node + nWaves) * DFEAT, 0, 0);

        // ---- issue bias loads early (latency hidden behind softmax)
        v4u bl[8], bh[8];
        asm volatile(
            "ds_load_b128 %0, %16\n\t"
            "ds_load_b128 %1, %16 offset:16\n\t"
            "ds_load_b128 %2, %16 offset:64\n\t"
            "ds_load_b128 %3, %16 offset:80\n\t"
            "ds_load_b128 %4, %16 offset:128\n\t"
            "ds_load_b128 %5, %16 offset:144\n\t"
            "ds_load_b128 %6, %16 offset:192\n\t"
            "ds_load_b128 %7, %16 offset:208\n\t"
            "ds_load_b128 %8, %16 offset:256\n\t"
            "ds_load_b128 %9, %16 offset:272\n\t"
            "ds_load_b128 %10, %16 offset:320\n\t"
            "ds_load_b128 %11, %16 offset:336\n\t"
            "ds_load_b128 %12, %16 offset:384\n\t"
            "ds_load_b128 %13, %16 offset:400\n\t"
            "ds_load_b128 %14, %16 offset:448\n\t"
            "ds_load_b128 %15, %16 offset:464"
            : "=v"(bl[0]), "=v"(bh[0]), "=v"(bl[1]), "=v"(bh[1]),
              "=v"(bl[2]), "=v"(bh[2]), "=v"(bl[3]), "=v"(bh[3]),
              "=v"(bl[4]), "=v"(bh[4]), "=v"(bl[5]), "=v"(bh[5]),
              "=v"(bl[6]), "=v"(bh[6]), "=v"(bl[7]), "=v"(bh[7])
            : "v"(b1lds));

        const int nbr = (int)indices[(long long)node * DEG + col];

        // ---- 2-D cosine score + 16-lane segment softmax (halves duplicated)
        const float q0 = y[node * DFEAT + 0], q1 = y[node * DFEAT + 1];
        const float k0 = y[nbr  * DFEAT + 0], k1 = y[nbr  * DFEAT + 1];
        const float rq = __builtin_amdgcn_rsqf(fmaxf(q0 * q0 + q1 * q1, 1e-18f));
        const float rk = __builtin_amdgcn_rsqf(fmaxf(k0 * k0 + k1 * k1, 1e-18f));
        float s = (q0 * k0 + q1 * k1) * rq * rk;
        float mx = s;
        #pragma unroll
        for (int off = 1; off < 16; off <<= 1) mx = fmaxf(mx, __shfl_xor(mx, off));
        float ex  = __builtin_amdgcn_exp2f((s - mx) * 1.44269504088896340736f);
        float sum = ex;
        #pragma unroll
        for (int off = 1; off < 16; off <<= 1) sum += __shfl_xor(sum, off);
        const float a = ex * __builtin_amdgcn_rcpf(sum);

        // ---- complete bias loads, seed accumulators
        asm volatile("s_wait_dscnt 0x0"
                     : "+v"(bl[0]), "+v"(bh[0]), "+v"(bl[1]), "+v"(bh[1]),
                       "+v"(bl[2]), "+v"(bh[2]), "+v"(bl[3]), "+v"(bh[3]),
                       "+v"(bl[4]), "+v"(bh[4]), "+v"(bl[5]), "+v"(bh[5]),
                       "+v"(bl[6]), "+v"(bh[6]), "+v"(bl[7]), "+v"(bh[7]));
        v8f acc[8];
        #pragma unroll
        for (int mt = 0; mt < 8; ++mt) acc[mt] = accf_from(bl[mt], bh[mt]);

        // ---- GEMM1: hT = W1^T @ aggT (register-resident A, streamed B)
        #pragma unroll
        for (int kt = 0; kt < 4; ++kt) {
            // B-fragment: aggT rows K = 32kt+8hi+0..7 and 32kt+16+8hi+0..7
            const int row = (kt < 2) ? nbr : node;
            const _Float16* src = y_h + (long long)row * DFEAT + (kt & 1) * 32 + 8 * hi;
            const v16h B = load_frag_gl(src);
            #pragma unroll
            for (int mt = 0; mt < 8; ++mt)
                acc[mt] = __builtin_amdgcn_wmma_f32_16x16x32_f16(
                    false, A1[mt * 4 + kt], false, B, (short)0, acc[mt], false, false);
        }

        // ---- relu + softmax scale fused into f16 pair packing.
        unsigned int pk[8][4];
        #pragma unroll
        for (int t = 0; t < 8; ++t)
            #pragma unroll
            for (int w = 0; w < 4; ++w)
                pk[t][w] = pack_rtz(fmaxf(acc[t][2 * w],     0.0f) * a,
                                    fmaxf(acc[t][2 * w + 1], 0.0f) * a);

        // ---- GEMM2: mT = W2^T @ hT ; hT -> B-fragments purely in registers
        v8f acc2[4];
        #pragma unroll
        for (int mt = 0; mt < 4; ++mt)
            #pragma unroll
            for (int v = 0; v < 8; ++v) acc2[mt][v] = 0.0f;

        #pragma unroll
        for (int kt = 0; kt < 4; ++kt) {
            // issue the 4 W2^T fragments for this K-step (mt stride 4096 B)
            v4u a2[8];
            const unsigned base = w2lds + (unsigned)((kt * 32 + lane) * 32);
            asm volatile(
                "ds_load_b128 %0, %8\n\t"
                "ds_load_b128 %1, %8 offset:16\n\t"
                "ds_load_b128 %2, %8 offset:4096\n\t"
                "ds_load_b128 %3, %8 offset:4112\n\t"
                "ds_load_b128 %4, %8 offset:8192\n\t"
                "ds_load_b128 %5, %8 offset:8208\n\t"
                "ds_load_b128 %6, %8 offset:12288\n\t"
                "ds_load_b128 %7, %8 offset:12304"
                : "=v"(a2[0]), "=v"(a2[1]), "=v"(a2[2]), "=v"(a2[3]),
                  "=v"(a2[4]), "=v"(a2[5]), "=v"(a2[6]), "=v"(a2[7])
                : "v"(base));

            // K-step kt consumes hT tiles 2kt,2kt+1; each B pair (j=2w,2w+1)
            // comes from this lane's own pk word: no cross-lane shuffle.
            FragU Bf;
            #pragma unroll
            for (int w = 0; w < 4; ++w) {
                const unsigned int pLo = pk[2 * kt][w];
                const unsigned int pHi = pk[2 * kt + 1][w];
                Bf.u[w]     = pLo;   // K offsets 8hi+2w, 8hi+2w+1
                Bf.u[4 + w] = pHi;   // K offsets 16+8hi+2w, 17+8hi+2w
            }

            asm volatile("s_wait_dscnt 0x0"
                         : "+v"(a2[0]), "+v"(a2[1]), "+v"(a2[2]), "+v"(a2[3]),
                           "+v"(a2[4]), "+v"(a2[5]), "+v"(a2[6]), "+v"(a2[7]));
            #pragma unroll
            for (int mt = 0; mt < 4; ++mt) {
                const v16h A = frag_from(a2[2 * mt], a2[2 * mt + 1]);
                acc2[mt] = __builtin_amdgcn_wmma_f32_16x16x32_f16(
                    false, A, false, Bf.h, (short)0, acc2[mt], false, false);
            }
        }

        // ---- reduce over the 16 edge-columns, add b2 (softmax sums to 1)
        #pragma unroll
        for (int mt = 0; mt < 4; ++mt) {
            float r[8];
            #pragma unroll
            for (int v = 0; v < 8; ++v) {
                float x = acc2[mt][v];
                #pragma unroll
                for (int off = 1; off < 16; off <<= 1) x += __shfl_xor(x, off);
                r[v] = x;
            }
            if (col == 0) {   // lanes 0 and 16 each own 8 contiguous features
                const int fb = 16 * mt + 8 * hi;
                float4 bA = *(const float4*)(b2 + fb);
                float4 bB = *(const float4*)(b2 + fb + 4);
                float4 oA = make_float4(r[0] + bA.x, r[1] + bA.y, r[2] + bA.z, r[3] + bA.w);
                float4 oB = make_float4(r[4] + bB.x, r[5] + bB.y, r[6] + bB.z, r[7] + bB.w);
                float* op = out + (long long)node * DFEAT + fb;
                *(float4*)(op)     = oA;
                *(float4*)(op + 4) = oB;
            }
        }
    }
}

// ---------------------------------------------------------------------------
extern "C" void kernel_launch(void* const* d_in, const int* in_sizes, int n_in,
                              void* d_out, int out_size, void* d_ws, size_t ws_size,
                              hipStream_t stream) {
    (void)n_in; (void)out_size; (void)ws_size;
    const float*     y       = (const float*)d_in[0];
    const long long* indices = (const long long*)d_in[1];
    // d_in[2] = indptr: fixed CSR (indptr[i] = i*16), not needed
    const float*     W1      = (const float*)d_in[3];
    const float*     b1      = (const float*)d_in[4];
    const float*     W2      = (const float*)d_in[5];
    const float*     b2      = (const float*)d_in[6];
    float*           out     = (float*)d_out;

    const int n_y     = in_sizes[0];        // N*64
    const int n_nodes = n_y / DFEAT;

    char* ws = (char*)d_ws;
    unsigned int* y_h = (unsigned int*)ws;                       // n_y f16 = n_y*2 B
    size_t off = (((size_t)n_y * 2) + 255) & ~(size_t)255;
    unsigned int* w1p = (unsigned int*)(ws + off);               // 32 KB
    unsigned int* w2p = (unsigned int*)(ws + off + 8192 * 4);    // 16 KB

    const int npairs = n_y / 2;
    k_cvt_y<<<(npairs + 255) / 256, 256, 0, stream>>>(y, y_h, npairs);
    k_pack_w<<<48, 256, 0, stream>>>(W1, W2, w1p, w2p);

    int blocks = 1024;
    if (blocks * 8 > n_nodes) blocks = (n_nodes + 7) / 8;
    k_agno<<<blocks, 256, 0, stream>>>(y, indices, b1, b2,
                                       (const _Float16*)y_h, w1p, w2p,
                                       out, n_nodes);
}
